// BasicBlockRetriever_51135880626362
// MI455X (gfx1250) — matrix operand order repaired and checked by reading, not compiled, not run on hardware
//
#include <hip/hip_runtime.h>
#include <math.h>
#include <stdint.h>

// ---------------------------------------------------------------------------
// Types / WMMA helpers (CDNA5 gfx1250, wave32, v_wmma_f32_16x16x32_bf16)
// ---------------------------------------------------------------------------
typedef __attribute__((ext_vector_type(16))) __bf16 v16bf;
typedef __attribute__((ext_vector_type(8)))  __bf16 v8bf;
typedef __attribute__((ext_vector_type(8)))  float  v8f;
typedef __attribute__((ext_vector_type(4)))  unsigned int u32x4;
typedef __attribute__((ext_vector_type(8)))  int i32x8;
typedef __attribute__((ext_vector_type(4)))  int i32x4;

static __device__ __forceinline__ __bf16 tobf(float f)   { return (__bf16)f; }
static __device__ __forceinline__ float  frombf(__bf16 h){ return (float)h; }

static __device__ __forceinline__ v8f zero8f() {
  v8f v;
#pragma unroll
  for (int i = 0; i < 8; ++i) v[i] = 0.0f;
  return v;
}
static __device__ __forceinline__ v16bf zero16bf() {
  v16bf v;
#pragma unroll
  for (int i = 0; i < 16; ++i) v[i] = (__bf16)0.0f;
  return v;
}

// A-fragment: lane-relative base already applied; loads K {+0..7} and {+8..15}
static __device__ __forceinline__ v16bf ldA(const __bf16* p) {
  v8bf lo = *(const v8bf*)(p);
  v8bf hi = *(const v8bf*)(p + 8);
  v16bf a;
#pragma unroll
  for (int i = 0; i < 8; ++i) { a[i] = lo[i]; a[i + 8] = hi[i]; }
  return a;
}

static __device__ __forceinline__ v8f wmma_bf16(v16bf a, v16bf b, v8f c) {
  return __builtin_amdgcn_wmma_f32_16x16x32_bf16(false, a, false, b,
                                                 (short)0, c, false, false);
}

static __device__ __forceinline__ float wred_sum(float v) {
#pragma unroll
  for (int m = 16; m >= 1; m >>= 1) v += __shfl_xor(v, m, 32);
  return v;
}
static __device__ __forceinline__ float wred_max(float v) {
#pragma unroll
  for (int m = 16; m >= 1; m >>= 1) v = fmaxf(v, __shfl_xor(v, m, 32));
  return v;
}

#define NBUF   100000
#define DDIM   266
#define EPSLN  1e-5f

#if __has_builtin(__builtin_amdgcn_tensor_load_to_lds) && \
    __has_builtin(__builtin_amdgcn_s_wait_tensorcnt)
#define USE_TDM 1
#else
#define USE_TDM 0
#endif

#if USE_TDM
// 2D TDM: load tile (tile_dim0=32 elems of 2B contiguous, tile_dim1=256 rows,
// row stride 256 elems) from global 'gptr' into LDS byte offset 'lds_off'.
// This toolchain exposes the 6-arg builtin:
//   (u32x4 g0, i32x8 g1, i32x4 g2, i32x4 g3, i32x8 g4, i32 cpol)
static __device__ __forceinline__ void tdm_load_tile(unsigned lds_off,
                                                     const __bf16* gptr) {
  unsigned long long ga = (unsigned long long)(uintptr_t)gptr;
  u32x4 g0;
  g0[0] = 1u;                                   // count=1 (valid user D#)
  g0[1] = lds_off;                              // D#.lds_addr (bytes)
  g0[2] = (unsigned)(ga & 0xffffffffu);         // global_addr[31:0]
  g0[3] = (unsigned)((ga >> 32) & 0x01ffffffu)  // global_addr[56:32]
          | (2u << 30);                         // type = 2 ("image")
  i32x8 g1;
  g1[0] = (int)(1u << 16);                      // data_size=1 -> 2 bytes
  g1[1] = (int)(256u << 16);                    // tensor_dim0 = 256 (lo16)
  g1[2] = (int)(256u << 16);                    // dim0 hi=0 | tensor_dim1 lo16=256
  g1[3] = (int)(32u << 16);                     // dim1 hi=0 | tile_dim0 = 32
  g1[4] = (int)256u;                            // tile_dim1=256 | tile_dim2=0
  g1[5] = (int)256u;                            // tensor_dim0_stride lo32 = 256
  g1[6] = 0;                                    // stride0 hi16 | stride1 lo16
  g1[7] = 0;                                    // stride1 hi32
  i32x4 z4;
  z4[0] = 0; z4[1] = 0; z4[2] = 0; z4[3] = 0;
  i32x8 z8;
#pragma unroll
  for (int i = 0; i < 8; ++i) z8[i] = 0;
  __builtin_amdgcn_tensor_load_to_lds(g0, g1, z4, z4, z8, 0);
}
#endif

// ---------------------------------------------------------------------------
// Prep kernels
// ---------------------------------------------------------------------------
__global__ __launch_bounds__(256) void xt_kernel(const float* __restrict__ x,
                                                 __bf16* __restrict__ xt) {
  int i = blockIdx.x * 256 + threadIdx.x;      // i = b*65536 + p*256 + c
  int b = i >> 16, p = (i >> 8) & 255, c = i & 255;
  xt[i] = tobf(x[(((size_t)b << 8) | c) << 8 | p]);
}

__global__ __launch_bounds__(256) void convw_kernel(const float* __restrict__ src,
                                                    __bf16* __restrict__ dst) {
  int i = blockIdx.x * 256 + threadIdx.x;
  if (i >= 256 * 256 * 9) return;
  int co = i / 2304, rem = i % 2304, ci = rem / 9, k9 = rem % 9;
  dst[((size_t)k9 * 256 + co) * 256 + ci] = tobf(src[i]);
}

__global__ __launch_bounds__(256) void f2bf_t_kernel(const float* __restrict__ src,
                                                     __bf16* __restrict__ dst,
                                                     int R, int N) {
  int i = blockIdx.x * 256 + threadIdx.x;
  if (i >= R * N) return;
  int r = i / N, n = i % N;
  dst[(size_t)n * R + r] = tobf(src[i]);
}

// ---------------------------------------------------------------------------
// Conv as 9 shifted WMMA GEMMs with TDM/async double-buffered LDS staging.
// Block = 256 thr (8 waves), one block per (batch, 64-c_out slab).
// Per ci-chunk (32 channels): the whole 256-position activation slice is
// staged once into LDS (16 KB) and reused by all 9 taps x 8 waves.
// mode 0: y = conv*g+b, relu -> bf16 token-major; mode 1: +x res, relu -> fp32
// ---------------------------------------------------------------------------
__global__ __launch_bounds__(256) void conv_wmma_kernel(
    const __bf16* __restrict__ xin, const __bf16* __restrict__ w9,
    const float* __restrict__ gam, const float* __restrict__ bet,
    const float* __restrict__ xres, float* __restrict__ outF,
    __bf16* __restrict__ outH, int mode) {
  __shared__ __bf16 sbuf[2][256 * 32];          // 2 x 16 KB
  int b = blockIdx.x >> 2, cb = blockIdx.x & 3;
  int wv = threadIdx.x >> 5, lane = threadIdx.x & 31;
  int mt = wv & 3, half = wv >> 2;
  int coBase = cb * 64 + mt * 16;
  int arow = coBase + (lane & 15);
  int aoff = (lane & 16) ? 16 : 0;
  int boff = (lane & 16) ? 16 : 0;
  const __bf16* xb = xin + ((size_t)b << 16);

  v8f acc[8];
#pragma unroll
  for (int t = 0; t < 8; ++t) acc[t] = zero8f();

  // --- staging helpers ---
  auto stage = [&](int kc, int buf) {
#if USE_TDM
    if (threadIdx.x < 32) {
      unsigned lds_off = (unsigned)(uintptr_t)(&sbuf[buf][0]);
      tdm_load_tile(lds_off, xb + kc * 32);
    }
#else
    // cooperative copy: 256 pos x 32 ch, 8 bf16 per transfer
    for (int i = threadIdx.x; i < 1024; i += 256) {
      int p = i >> 2, g = i & 3;
      *(v8bf*)(&sbuf[buf][p * 32 + g * 8]) =
          *(const v8bf*)(xb + (size_t)p * 256 + kc * 32 + g * 8);
    }
#endif
  };
  auto wait_stage = [&]() {
#if USE_TDM
    if (threadIdx.x < 32) __builtin_amdgcn_s_wait_tensorcnt(0);
#endif
  };

  stage(0, 0);
  wait_stage();
  __syncthreads();

  for (int kc = 0; kc < 8; ++kc) {
    int k0 = kc * 32;
    int cur = kc & 1;
    if (kc < 7) stage(kc + 1, cur ^ 1);         // overlap next chunk load
    const __bf16* lb = &sbuf[cur][0];
    for (int k9 = 0; k9 < 9; ++k9) {
      int dy = k9 / 3 - 1, dx = k9 % 3 - 1;
      v16bf a = ldA(w9 + ((size_t)k9 << 16) + (size_t)arow * 256 + aoff + k0);
#pragma unroll
      for (int t = 0; t < 8; ++t) {
        int p = ((half << 3) + t) * 16 + (lane & 15);
        int y = p >> 4, x = p & 15;
        int yy = y + dy, xx = x + dx;
        v16bf bf = zero16bf();
        if (yy >= 0 && yy < 16 && xx >= 0 && xx < 16)
          bf = *(const v16bf*)(lb + (size_t)((yy << 4) + xx) * 32 + boff);
        acc[t] = wmma_bf16(a, bf, acc[t]);
      }
    }
    if (kc < 7) wait_stage();
    __syncthreads();
  }

  int rbase = (lane & 16) ? 8 : 0;
#pragma unroll
  for (int t = 0; t < 8; ++t) {
    int p = ((half << 3) + t) * 16 + (lane & 15);
#pragma unroll
    for (int r = 0; r < 8; ++r) {
      int c = coBase + rbase + r;
      float v = acc[t][r] * gam[c] + bet[c];
      if (mode == 0) {
        v = fmaxf(v, 0.0f);
        outH[((((size_t)b << 8) | p) << 8) | c] = tobf(v);
      } else {
        v += xres[((((size_t)b << 8) | c) << 8) | p];
        v = fmaxf(v, 0.0f);
        outF[((((size_t)b << 8) | p) << 8) | c] = v;
      }
    }
  }
}

// ---------------------------------------------------------------------------
// Generic WMMA GEMM: C = A(MxK bf16,row-major) * Bt(NxK bf16)^T
// mode: 0 store bf16   1 fp32*scale   2 fp32+bias[col]
//       3 fp32+bias+res   4 fp32+bias+res scattered to NCHW (final output)
// bStride: per-batch Bt stride (batch = m0/256), 0 if shared B.
// ---------------------------------------------------------------------------
template <int NACC>
__global__ __launch_bounds__(256) void gemm_wmma_kernel(
    const __bf16* __restrict__ A, int lda, const __bf16* __restrict__ Bt,
    int ldb, int M, int N, int K, int mode, float scale,
    const float* __restrict__ bias, const float* __restrict__ res,
    float* __restrict__ outF, __bf16* __restrict__ outH, int bStride) {
  int wave = blockIdx.x * 8 + (threadIdx.x >> 5);
  int lane = threadIdx.x & 31;
  int numMT = M >> 4;
  const int span = NACC * 16;
  int total = numMT * (N / span);
  if (wave >= total) return;
  int mt = wave % numMT, nb = wave / numMT;
  int m0 = mt << 4, n0 = nb * span;
  const __bf16* Bb = Bt + (bStride ? (size_t)(m0 >> 8) * (size_t)bStride : 0);
  v8f acc[NACC];
#pragma unroll
  for (int t = 0; t < NACC; ++t) acc[t] = zero8f();
  int aoff = (lane & 16) ? 16 : 0, boff = (lane & 16) ? 16 : 0;
  const __bf16* ap = A + (size_t)(m0 + (lane & 15)) * lda + aoff;
  for (int k0 = 0; k0 < K; k0 += 32) {
    v16bf a = ldA(ap + k0);
#pragma unroll
    for (int t = 0; t < NACC; ++t) {
      const __bf16* bp = Bb + (size_t)(n0 + t * 16 + (lane & 15)) * ldb + k0 + boff;
      acc[t] = wmma_bf16(a, *(const v16bf*)bp, acc[t]);
    }
  }
  int rbase = m0 + ((lane & 16) ? 8 : 0);
#pragma unroll
  for (int t = 0; t < NACC; ++t) {
    int col = n0 + t * 16 + (lane & 15);
#pragma unroll
    for (int r = 0; r < 8; ++r) {
      int row = rbase + r;
      float v = acc[t][r];
      if (mode == 0)      outH[(size_t)row * N + col] = tobf(v);
      else if (mode == 1) outF[(size_t)row * N + col] = v * scale;
      else if (mode == 2) outF[(size_t)row * N + col] = v + bias[col];
      else if (mode == 3) outF[(size_t)row * N + col] = v + bias[col] + res[(size_t)row * N + col];
      else {
        int bb2 = row >> 8, p = row & 255;
        outF[((((size_t)bb2 << 8) | (unsigned)col) << 8) | p] =
            v + bias[col] + res[(size_t)row * 256 + col];
      }
    }
  }
}

// ---------------------------------------------------------------------------
// LayerNorm over 256 channels per token row: dst = bf16(LN(src)*g+b)
// ---------------------------------------------------------------------------
__global__ __launch_bounds__(256) void ln_rows_kernel(
    const float* __restrict__ src, const float* __restrict__ g,
    const float* __restrict__ b, __bf16* __restrict__ dst) {
  int row = blockIdx.x, tid = threadIdx.x;
  float v = src[(size_t)row * 256 + tid];
  __shared__ float s1[256], s2[256];
  s1[tid] = v; s2[tid] = v * v;
  __syncthreads();
  for (int o = 128; o > 0; o >>= 1) {
    if (tid < o) { s1[tid] += s1[tid + o]; s2[tid] += s2[tid + o]; }
    __syncthreads();
  }
  float m = s1[0] * (1.0f / 256.0f);
  float var = s2[0] * (1.0f / 256.0f) - m * m;
  float rs = rsqrtf(var + EPSLN);
  dst[(size_t)row * 256 + tid] = tobf((v - m) * rs * g[tid] + b[tid]);
}

// e0[b][r] = sum_d q_bf16[b*256+0][d] * wqe[d][r]
__global__ __launch_bounds__(256) void e0_kernel(const __bf16* __restrict__ q,
                                                 const float* __restrict__ wqe,
                                                 float* __restrict__ e0) {
  int b = blockIdx.x, r = threadIdx.x;
  const __bf16* q0 = q + (size_t)b * 256 * 64;
  float s = 0.0f;
  for (int d = 0; d < 64; ++d) s += frombf(q0[d]) * wqe[d * 256 + r];
  e0[b * 256 + r] = s;
}

// ---------------------------------------------------------------------------
// Fused ctx-LN + L2 distance: one wave per retrieval row; e0 in LDS with
// stride-257 padding (no 32-way bank conflicts). dist[q][row].
// ---------------------------------------------------------------------------
__global__ __launch_bounds__(256) void dist_kernel(
    const float* __restrict__ rd, const float* __restrict__ gg,
    const float* __restrict__ bb, const float* __restrict__ e0,
    float* __restrict__ dist) {
  __shared__ float e0s[32 * 257];
  __shared__ float gl[256], bl[256];
  int tid = threadIdx.x;
  for (int i = tid; i < 8192; i += 256) {
    int q = i >> 8, c = i & 255;
    e0s[q * 257 + c] = e0[i];
  }
  gl[tid] = gg[tid]; bl[tid] = bb[tid];
  __syncthreads();
  int wv = tid >> 5, lane = tid & 31;
  int row = blockIdx.x * 8 + wv;
  if (row >= NBUF) return;
  const float* r = rd + (size_t)row * DDIM;
  float vals[9]; float s = 0.0f, s2 = 0.0f;
#pragma unroll
  for (int i = 0; i < 9; ++i) {
    int c = lane + 32 * i;
    float x = (c < DDIM) ? r[c] : 0.0f;
    vals[i] = x; s += x; s2 += x * x;
  }
  s = wred_sum(s); s2 = wred_sum(s2);
  float m = s * (1.0f / DDIM);
  float var = s2 * (1.0f / DDIM) - m * m;
  float rs = rsqrtf(var + EPSLN);
  float d = 0.0f;
  const float* eq = e0s + lane * 257;
#pragma unroll
  for (int i = 0; i < 8; ++i) {
    float base = vals[i];
    for (int srcl = 0; srcl < 32; ++srcl) {
      float xv = __shfl(base, srcl, 32);
      int c = i * 32 + srcl;
      float cn = (xv - m) * rs * gl[c] + bl[c];
      float t = cn - eq[c];
      d += t * t;
    }
  }
  dist[(size_t)lane * NBUF + row] = d;
}

// Iterative top-32 argmin per batch (order irrelevant: softmax-gather is
// permutation invariant in the final output)
__global__ __launch_bounds__(1024) void topk_kernel(float* __restrict__ dist,
                                                    int* __restrict__ idxout) {
  int q = blockIdx.x, tid = threadIdx.x;
  __shared__ float bv[1024];
  __shared__ int   bi[1024];
  float* dq = dist + (size_t)q * NBUF;
  for (int it = 0; it < 32; ++it) {
    float best = 3.0e38f; int besti = NBUF;
    for (int r = tid; r < NBUF; r += 1024) {
      float v = dq[r];
      if (v < best || (v == best && r < besti)) { best = v; besti = r; }
    }
    bv[tid] = best; bi[tid] = besti;
    __syncthreads();
    for (int o = 512; o > 0; o >>= 1) {
      if (tid < o) {
        if (bv[tid + o] < bv[tid] ||
            (bv[tid + o] == bv[tid] && bi[tid + o] < bi[tid])) {
          bv[tid] = bv[tid + o]; bi[tid] = bi[tid + o];
        }
      }
      __syncthreads();
    }
    if (tid == 0) { idxout[q * 32 + it] = bi[0]; dq[bi[0]] = 3.0e38f; }
    __syncthreads();
  }
}

// Gather selected rows, re-LN, project: kb[b][j][d] = reps@wk (bf16),
// vt[b][d][j] = labels@wv (bf16, transposed for B-frag loads). One wave/(b,j).
__global__ __launch_bounds__(256) void gather_kv_kernel(
    const float* __restrict__ rd, const float* __restrict__ gg,
    const float* __restrict__ bb, const int* __restrict__ idx,
    const float* __restrict__ wk, const float* __restrict__ wv,
    __bf16* __restrict__ kb, __bf16* __restrict__ vt) {
  int wave = (blockIdx.x * 256 + threadIdx.x) >> 5;
  int lane = threadIdx.x & 31;
  int b = wave >> 5, j = wave & 31;
  int row = idx[b * 32 + j];
  const float* r = rd + (size_t)row * DDIM;
  float vals[9]; float s = 0.0f, s2 = 0.0f;
#pragma unroll
  for (int i = 0; i < 9; ++i) {
    int c = lane + 32 * i;
    float x = (c < DDIM) ? r[c] : 0.0f;
    vals[i] = x; s += x; s2 += x * x;
  }
  s = wred_sum(s); s2 = wred_sum(s2);
  float m = s * (1.0f / DDIM);
  float var = s2 * (1.0f / DDIM) - m * m;
  float rs = rsqrtf(var + EPSLN);
  float k0a = 0.0f, k1a = 0.0f;
#pragma unroll
  for (int i = 0; i < 8; ++i) {
    float base = vals[i];
    for (int srcl = 0; srcl < 32; ++srcl) {
      float xv = __shfl(base, srcl, 32);
      int c = i * 32 + srcl;
      float cn = (xv - m) * rs * gg[c] + bb[c];
      k0a += cn * wk[c * 64 + lane];
      k1a += cn * wk[c * 64 + lane + 32];
    }
  }
  kb[((size_t)(b * 32 + j)) * 64 + lane]      = tobf(k0a);
  kb[((size_t)(b * 32 + j)) * 64 + lane + 32] = tobf(k1a);
  float v0 = 0.0f, v1 = 0.0f;
  {
    float base = vals[8];  // c = lane + 256 (valid for lane < 10)
    for (int li = 0; li < 10; ++li) {
      float xv = __shfl(base, li, 32);
      int c = 256 + li;
      float cn = (xv - m) * rs * gg[c] + bb[c];
      v0 += cn * wv[li * 64 + lane];
      v1 += cn * wv[li * 64 + lane + 32];
    }
  }
  vt[((size_t)b * 64 + lane) * 32 + j]      = tobf(v0);
  vt[((size_t)b * 64 + lane + 32) * 32 + j] = tobf(v1);
}

// Softmax over 32 attention logits; one wave per token row.
__global__ __launch_bounds__(256) void softmax_kernel(
    const float* __restrict__ sim, __bf16* __restrict__ attn) {
  int wave = blockIdx.x * 8 + (threadIdx.x >> 5);
  int lane = threadIdx.x & 31;
  float v = sim[(size_t)wave * 32 + lane];
  float mx = wred_max(v);
  float e = expf(v - mx);
  float s = wred_sum(e);
  attn[(size_t)wave * 32 + lane] = tobf(e / s);
}

// act = a * gelu_exact(g), h = [a | g] (8192 x 512)
__global__ __launch_bounds__(256) void gate_kernel(const float* __restrict__ h,
                                                   __bf16* __restrict__ act) {
  int m = blockIdx.x, n = threadIdx.x;
  float a = h[(size_t)m * 512 + n];
  float g = h[(size_t)m * 512 + 256 + n];
  float ge = 0.5f * g * (1.0f + erff(g * 0.70710678118654752f));
  act[(size_t)m * 256 + n] = tobf(a * ge);
}

// ---------------------------------------------------------------------------
// Host orchestration
// ---------------------------------------------------------------------------
extern "C" void kernel_launch(void* const* d_in, const int* in_sizes, int n_in,
                              void* d_out, int out_size, void* d_ws,
                              size_t ws_size, hipStream_t stream) {
  (void)in_sizes; (void)n_in; (void)out_size; (void)ws_size;
  const float* x    = (const float*)d_in[0];
  const float* c1w  = (const float*)d_in[1];
  const float* c2w  = (const float*)d_in[2];
  const float* bn1g = (const float*)d_in[3];
  const float* bn1b = (const float*)d_in[4];
  const float* bn2g = (const float*)d_in[5];
  const float* bn2b = (const float*)d_in[6];
  const float* rd   = (const float*)d_in[7];
  const float* lag  = (const float*)d_in[8];
  const float* labp = (const float*)d_in[9];
  const float* lcg  = (const float*)d_in[10];
  const float* lcb  = (const float*)d_in[11];
  const float* wq   = (const float*)d_in[12];
  const float* wk   = (const float*)d_in[13];
  const float* wv   = (const float*)d_in[14];
  const float* wqe  = (const float*)d_in[15];
  const float* wo   = (const float*)d_in[16];
  const float* bo   = (const float*)d_in[17];
  const float* lfg  = (const float*)d_in[18];
  const float* lfb  = (const float*)d_in[19];
  const float* w1   = (const float*)d_in[20];
  const float* b1   = (const float*)d_in[21];
  const float* w2   = (const float*)d_in[22];
  const float* b2   = (const float*)d_in[23];
  float* out = (float*)d_out;

  char* wsp = (char*)d_ws;
  auto alloc = [&](size_t bytes) -> char* {
    char* p = wsp;
    wsp += (bytes + 255) & ~(size_t)255;
    return p;
  };
  __bf16* XT   = (__bf16*)alloc((size_t)2097152 * 2);     // x token-major bf16
  __bf16* WC1  = (__bf16*)alloc((size_t)589824 * 2);      // conv1 w (9,co,ci)
  __bf16* WC2  = (__bf16*)alloc((size_t)589824 * 2);
  __bf16* WQT  = (__bf16*)alloc((size_t)64 * 256 * 2);    // wq^T  (64 x 256)
  __bf16* WOT  = (__bf16*)alloc((size_t)256 * 64 * 2);    // wo^T  (256 x 64)
  __bf16* W1T  = (__bf16*)alloc((size_t)512 * 256 * 2);   // w1^T  (512 x 256)
  __bf16* W2T  = (__bf16*)alloc((size_t)256 * 256 * 2);   // w2^T  (256 x 256)
  __bf16* ACT1 = (__bf16*)alloc((size_t)2097152 * 2);     // conv1 out, token-major
  float*  T    = (float*)alloc((size_t)2097152 * 4);      // t (8192 x 256)
  __bf16* XN   = (__bf16*)alloc((size_t)2097152 * 2);     // LN(t) bf16
  __bf16* QB   = (__bf16*)alloc((size_t)8192 * 64 * 2);   // q bf16
  float*  E0   = (float*)alloc((size_t)32 * 256 * 4);
  float*  DIST = (float*)alloc((size_t)32 * NBUF * 4);
  int*    IDX  = (int*)alloc((size_t)1024 * 4);
  __bf16* KB   = (__bf16*)alloc((size_t)32 * 32 * 64 * 2);
  __bf16* VT   = (__bf16*)alloc((size_t)32 * 64 * 32 * 2);
  float*  SIM  = (float*)alloc((size_t)8192 * 32 * 4);
  __bf16* ATTN = (__bf16*)alloc((size_t)8192 * 32 * 2);
  __bf16* OB   = (__bf16*)alloc((size_t)8192 * 64 * 2);
  float*  T2   = (float*)alloc((size_t)2097152 * 4);
  __bf16* LN2  = (__bf16*)alloc((size_t)2097152 * 2);
  float*  HB   = (float*)alloc((size_t)8192 * 512 * 4);
  __bf16* ACTG = (__bf16*)alloc((size_t)2097152 * 2);

  // --- prep ---
  xt_kernel<<<8192, 256, 0, stream>>>(x, XT);
  convw_kernel<<<2304, 256, 0, stream>>>(c1w, WC1);
  convw_kernel<<<2304, 256, 0, stream>>>(c2w, WC2);
  f2bf_t_kernel<<<64, 256, 0, stream>>>(wq, WQT, 256, 64);
  f2bf_t_kernel<<<64, 256, 0, stream>>>(wo, WOT, 64, 256);
  f2bf_t_kernel<<<512, 256, 0, stream>>>(w1, W1T, 256, 512);
  f2bf_t_kernel<<<256, 256, 0, stream>>>(w2, W2T, 256, 256);

  // --- basic block (two WMMA convs, TDM-staged LDS double buffering) ---
  conv_wmma_kernel<<<128, 256, 0, stream>>>(XT, WC1, bn1g, bn1b, nullptr,
                                            nullptr, ACT1, 0);
  conv_wmma_kernel<<<128, 256, 0, stream>>>(ACT1, WC2, bn2g, bn2b, x, T,
                                            nullptr, 1);

  // --- attention front-end ---
  ln_rows_kernel<<<8192, 256, 0, stream>>>(T, lag, labp, XN);
  gemm_wmma_kernel<4><<<64, 256, 0, stream>>>(XN, 256, WQT, 256, 8192, 64, 256,
                                              0, 1.0f, nullptr, nullptr,
                                              nullptr, QB, 0);
  e0_kernel<<<32, 256, 0, stream>>>(QB, wqe, E0);

  // --- retrieval: fused LN+distance, top-32, gather+project ---
  dist_kernel<<<(NBUF + 7) / 8, 256, 0, stream>>>(rd, lcg, lcb, E0, DIST);
  topk_kernel<<<32, 1024, 0, stream>>>(DIST, IDX);
  gather_kv_kernel<<<128, 256, 0, stream>>>(rd, lcg, lcb, IDX, wk, wv, KB, VT);

  // --- attention core (batched WMMA GEMMs) ---
  gemm_wmma_kernel<2><<<64, 256, 0, stream>>>(QB, 64, KB, 64, 8192, 32, 64, 1,
                                              0.125f, nullptr, nullptr, SIM,
                                              nullptr, 32 * 64);
  softmax_kernel<<<1024, 256, 0, stream>>>(SIM, ATTN);
  gemm_wmma_kernel<4><<<64, 256, 0, stream>>>(ATTN, 32, VT, 32, 8192, 64, 32,
                                              0, 1.0f, nullptr, nullptr,
                                              nullptr, OB, 64 * 32);
  gemm_wmma_kernel<4><<<256, 256, 0, stream>>>(OB, 64, WOT, 64, 8192, 256, 64,
                                               3, 1.0f, bo, T, T2, nullptr, 0);

  // --- gated MLP + fused output transpose ---
  ln_rows_kernel<<<8192, 256, 0, stream>>>(T2, lfg, lfb, LN2);
  gemm_wmma_kernel<4><<<512, 256, 0, stream>>>(LN2, 256, W1T, 256, 8192, 512,
                                               256, 2, 1.0f, b1, nullptr, HB,
                                               nullptr, 0);
  gate_kernel<<<8192, 256, 0, stream>>>(HB, ACTG);
  gemm_wmma_kernel<4><<<256, 256, 0, stream>>>(ACTG, 256, W2T, 256, 8192, 256,
                                               256, 4, 1.0f, b2, T2, out,
                                               nullptr, 0);
}
